// GruDecoder_61770219651427
// MI455X (gfx1250) — matrix operand order repaired and checked by reading, not compiled, not run on hardware
//
#include <hip/hip_runtime.h>

// Problem constants (from the reference)
#define T_STEPS 512
#define BATCH   128
#define ESEQ    768
#define CONDW   256
#define HID     1024
#define KIN     1024          // COND + E_SEQ
#define G3      3072          // 3*H
#define MTOT    (T_STEPS * BATCH)
#define NPART   4             // split-K factor for the recurrent GEMM

// GEMM tiling
#define BM  128
#define BK  32
#define LDT 40                // padded LDS row length (elements) to spread banks

typedef float v8f  __attribute__((ext_vector_type(8)));
typedef __bf16 v16bf __attribute__((ext_vector_type(16)));
typedef unsigned int v4u __attribute__((ext_vector_type(4)));
typedef int v4i __attribute__((vector_size(4 * sizeof(int))));

struct V4U2 { v4u lo, hi; };
union FragBF { v16bf bf; V4U2 u; };

#define AS1 __attribute__((address_space(1)))
#define AS3 __attribute__((address_space(3)))

#if __has_builtin(__builtin_amdgcn_global_load_async_to_lds_b128) && \
    __has_builtin(__builtin_amdgcn_s_wait_asynccnt)
#define HAVE_ASYNC_LDS 1
#else
#define HAVE_ASYNC_LDS 0
#endif

__device__ __forceinline__ unsigned short f32_to_bf16(float f) {
    union { float f; unsigned int u; } x; x.f = f;
    unsigned int u = x.u;
    unsigned int r = u + 0x7FFFu + ((u >> 16) & 1u);      // round-to-nearest-even
    if ((u & 0x7F800000u) == 0x7F800000u) r = u;          // preserve inf/NaN
    return (unsigned short)(r >> 16);
}

// Stage one 16-byte chunk global -> LDS (async DMA path on gfx1250: tracked by
// ASYNCcnt, no VGPR round-trip; fallback is classic load/store staging).
__device__ __forceinline__ void stage16(const unsigned short* g, unsigned short* s) {
#if HAVE_ASYNC_LDS
    __builtin_amdgcn_global_load_async_to_lds_b128((AS1 v4i*)g, (AS3 v4i*)s, 0, 0);
#else
    *(v4u*)s = *(const v4u*)g;
#endif
}

__device__ __forceinline__ void stage_fence() {
#if HAVE_ASYNC_LDS
    __builtin_amdgcn_s_wait_asynccnt(0);
#endif
}

// ---- Convert both weight matrices f32 -> bf16 (row-major [N=3H][K] kept) ----
__global__ __launch_bounds__(256) void k_convert_weights(
    const float* __restrict__ Wih, const float* __restrict__ Whh,
    unsigned short* __restrict__ WihB, unsigned short* __restrict__ WhhB) {
    const int n = G3 * KIN;
    for (int i = blockIdx.x * blockDim.x + threadIdx.x; i < n;
         i += gridDim.x * blockDim.x) {
        WihB[i] = f32_to_bf16(Wih[i]);
        WhhB[i] = f32_to_bf16(Whh[i]);
    }
}

// ---- Build x = concat(word_emb, seq_emb) per (t,b), as bf16 [T*B][KIN] ----
__global__ __launch_bounds__(256) void k_build_x(
    const float* __restrict__ seq, const float* __restrict__ word,
    unsigned short* __restrict__ x) {
    const long n = (long)MTOT * KIN;
    for (long i = (long)blockIdx.x * blockDim.x + threadIdx.x; i < n;
         i += (long)gridDim.x * blockDim.x) {
        int  k = (int)(i % KIN);
        long m = i / KIN;
        int  b = (int)(m % BATCH);
        long t = m / BATCH;
        float v = (k < CONDW) ? word[b * CONDW + k]
                              : seq[(t * BATCH + b) * (long)ESEQ + (k - CONDW)];
        x[i] = f32_to_bf16(v);
    }
}

// ---- Initialize hidden state (f32 carry + bf16 operand copy) ----
__global__ __launch_bounds__(256) void k_init_h(
    const float* __restrict__ h0, float* __restrict__ h,
    unsigned short* __restrict__ hb) {
    int i = blockIdx.x * blockDim.x + threadIdx.x;
    if (i < BATCH * HID) {
        float v = h0[i];
        h[i]  = v;
        hb[i] = f32_to_bf16(v);
    }
}

// ---- bf16 WMMA GEMM (no bias): C_part[M][N] = A[M][K-chunk] * W[N][K-chunk]^T
// A row-major bf16, W row-major bf16 ([N][K] == B-matrix column data), C f32.
// Block = 256 threads = 8 waves; block tile 128 x BN_T, K-step 32.
// Waves: 4 along M (32 rows) x 2 along N (BN_T/2 cols) => 2 x (BN_T/32) accs.
// blockIdx.z selects a K chunk of length K/kSplit; partial written to
// C + z*M*N (consumer sums the partials).
template <int BN_T>
__global__ __launch_bounds__(256) void k_gemm_bf16(
    const unsigned short* __restrict__ A,
    const unsigned short* __restrict__ W,
    float* __restrict__ C,
    int M, int N, int K, int kSplit) {
    constexpr int ACC_N = BN_T / 32;
    __shared__ unsigned short sA[BM * LDT];
    __shared__ unsigned short sB[BN_T * LDT];

    const int tid  = threadIdx.x;
    const int lane = tid & 31;
    const int wave = tid >> 5;
    const int wm   = wave & 3;   // 32-row strip within block tile
    const int wn   = wave >> 2;  // (BN_T/2)-col strip within block tile

    const long m0 = (long)blockIdx.y * BM;
    const long n0 = (long)blockIdx.x * BN_T;

    const int  kLen   = K / kSplit;
    const int  kStart = blockIdx.z * kLen;
    const int  kEnd   = kStart + kLen;
    float* Cp = C + (size_t)blockIdx.z * (size_t)M * N;

    v8f acc[2][ACC_N];
#pragma unroll
    for (int i = 0; i < 2; ++i)
#pragma unroll
        for (int j = 0; j < ACC_N; ++j) { v8f z = {}; acc[i][j] = z; }

    const int lr = tid >> 2;        // staging: row 0..63 per pass
    const int lc = (tid & 3) * 8;   // staging: 8-element (16B) chunk

    for (int k0 = kStart; k0 < kEnd; k0 += BK) {
        // Cooperative stage of A and W tiles into LDS
#pragma unroll
        for (int rr = 0; rr < BM; rr += 64) {
            const int row = lr + rr;
            stage16(&A[(m0 + row) * (long)K + k0 + lc], &sA[row * LDT + lc]);
        }
#pragma unroll
        for (int rr = 0; rr < BN_T; rr += 64) {
            const int row = lr + rr;
            stage16(&W[(n0 + row) * (long)K + k0 + lc], &sB[row * LDT + lc]);
        }
        if (k0 + BK < kEnd) {  // gfx1250 global_prefetch_b8
            __builtin_prefetch(&A[(m0 + lr) * (long)K + k0 + BK + lc], 0, 3);
            __builtin_prefetch(&W[(n0 + lr) * (long)K + k0 + BK + lc], 0, 3);
        }
        stage_fence();
        __syncthreads();

        // A fragments: 16x32 bf16. Lane half 0 -> K {0..7,16..23}, half 1 -> +8.
        FragBF a[2];
        {
            const int r  = lane & 15;
            const int kb = (lane >> 4) << 3;   // 0 or 8
#pragma unroll
            for (int i = 0; i < 2; ++i) {
                const unsigned short* p = &sA[(wm * 32 + i * 16 + r) * LDT + kb];
                a[i].u.lo = *(const v4u*)p;          // K = kb .. kb+7
                a[i].u.hi = *(const v4u*)(p + 16);   // K = kb+16 .. kb+23
            }
        }
        // B fragments: 32x16 bf16. Lane half 0 -> K {0..7,16..23}, half 1 -> +8.
        FragBF bq[ACC_N];
        {
            const int cidx = lane & 15;
            const int kb   = (lane >> 4) << 3;  // 0 or 8
#pragma unroll
            for (int j = 0; j < ACC_N; ++j) {
                const unsigned short* p =
                    &sB[(wn * (BN_T / 2) + j * 16 + cidx) * LDT + kb];
                bq[j].u.lo = *(const v4u*)p;        // K = kb .. kb+7
                bq[j].u.hi = *(const v4u*)(p + 16); // K = kb+16 .. kb+23
            }
        }
#pragma unroll
        for (int i = 0; i < 2; ++i)
#pragma unroll
            for (int j = 0; j < ACC_N; ++j)
                acc[i][j] = __builtin_amdgcn_wmma_f32_16x16x32_bf16(
                    false, a[i].bf, false, bq[j].bf, (short)0, acc[i][j],
                    false, false);
        __syncthreads();
    }

    // Epilogue: C/D layout — VGPR r: M = r + 8*(lane>>4), N = lane&15.
    const int rowHalf = (lane >> 4) * 8;
    const int colIn   = lane & 15;
#pragma unroll
    for (int j = 0; j < ACC_N; ++j) {
        const long col = n0 + wn * (BN_T / 2) + j * 16 + colIn;
#pragma unroll
        for (int i = 0; i < 2; ++i) {
            const long rowBase = m0 + wm * 32 + i * 16 + rowHalf;
#pragma unroll
            for (int r = 0; r < 8; ++r)
                Cp[(rowBase + r) * (long)N + col] = acc[i][j][r];
        }
    }
}

// ---- Fused GRU gate math: sums split-K partials of gh, adds both biases,
// updates hidden state (+ bf16 recast for next step's GEMM) ----
__global__ __launch_bounds__(256) void k_gru_update(
    const float* __restrict__ gx_t,   // [B][3H] for this timestep (no bias)
    const float* __restrict__ ghp,    // NPART x [B][3H] partials (no bias)
    const float* __restrict__ bih,    // [3H]
    const float* __restrict__ bhh,    // [3H]
    float* __restrict__ h,            // [B][H] in/out (f32 carry)
    unsigned short* __restrict__ hb,  // [B][H] bf16 operand for next step
    float* __restrict__ out_t) {      // d_out + t*B*H
    const int i = blockIdx.x * blockDim.x + threadIdx.x;
    if (i >= BATCH * HID) return;
    const int b = i / HID, c = i % HID;
    const float* gxr = gx_t + (long)b * G3;
    float xr = gxr[c] + bih[c];
    float xz = gxr[HID + c] + bih[HID + c];
    float xn = gxr[2 * HID + c] + bih[2 * HID + c];
    float hr = bhh[c], hz = bhh[HID + c], hn = bhh[2 * HID + c];
#pragma unroll
    for (int s = 0; s < NPART; ++s) {
        const float* p = ghp + (size_t)s * BATCH * G3 + (long)b * G3;
        hr += p[c];
        hz += p[HID + c];
        hn += p[2 * HID + c];
    }
    const float r = 1.0f / (1.0f + __expf(-(xr + hr)));
    const float z = 1.0f / (1.0f + __expf(-(xz + hz)));
    const float n = tanhf(xn + r * hn);
    const float hnew = (1.0f - z) * n + z * h[i];
    h[i]  = hnew;
    hb[i] = f32_to_bf16(hnew);
    out_t[i] = hnew;
}

extern "C" void kernel_launch(void* const* d_in, const int* in_sizes, int n_in,
                              void* d_out, int out_size, void* d_ws, size_t ws_size,
                              hipStream_t stream) {
    const float* seq  = (const float*)d_in[0];  // [T,B,768]
    const float* word = (const float*)d_in[1];  // [B,256]
    const float* h0   = (const float*)d_in[2];  // [1,B,H]
    const float* Wih  = (const float*)d_in[3];  // [3H, 1024]
    const float* Whh  = (const float*)d_in[4];  // [3H, H]
    const float* bih  = (const float*)d_in[5];  // [3H]
    const float* bhh  = (const float*)d_in[6];  // [3H]
    float* out = (float*)d_out;

    // Workspace carve-up (~950 MB)
    char* ws = (char*)d_ws;
    size_t off = 0;
    auto carve = [&](size_t bytes) -> char* {
        char* p = ws + off;
        off += (bytes + 255) & ~(size_t)255;
        return p;
    };
    unsigned short* xcat = (unsigned short*)carve((size_t)MTOT * KIN * 2);
    unsigned short* WihB = (unsigned short*)carve((size_t)G3 * KIN * 2);
    unsigned short* WhhB = (unsigned short*)carve((size_t)G3 * HID * 2);
    float*          gx   = (float*)carve((size_t)MTOT * G3 * 4);
    float*          ghp  = (float*)carve((size_t)NPART * BATCH * G3 * 4);
    float*          h    = (float*)carve((size_t)BATCH * HID * 4);
    unsigned short* hb   = (unsigned short*)carve((size_t)BATCH * HID * 2);

    // Phase 0: operand prep
    k_convert_weights<<<1024, 256, 0, stream>>>(Wih, Whh, WihB, WhhB);
    k_build_x<<<4096, 256, 0, stream>>>(seq, word, xcat);
    k_init_h<<<(BATCH * HID + 255) / 256, 256, 0, stream>>>(h0, h, hb);

    // Phase 1: gx = x @ W_ih^T   (65536 x 3072 x 1024, bf16 WMMA, no split)
    dim3 g1(G3 / 128, MTOT / BM, 1);  // (24, 512, 1)
    k_gemm_bf16<128><<<g1, 256, 0, stream>>>(xcat, WihB, gx, MTOT, G3, KIN, 1);

    // Phase 2: sequential GRU scan. Per step: gh partials via BN=64 tiles with
    // 4-way split-K (48*4 = 192 workgroups), then fused gate/update kernel.
    dim3 g2(G3 / 64, BATCH / BM, NPART); // (48, 1, 4)
    for (int t = 0; t < T_STEPS; ++t) {
        k_gemm_bf16<64><<<g2, 256, 0, stream>>>(hb, WhhB, ghp, BATCH, G3, HID,
                                                NPART);
        k_gru_update<<<(BATCH * HID + 255) / 256, 256, 0, stream>>>(
            gx + (size_t)t * BATCH * G3, ghp, bih, bhh, h, hb,
            out + (size_t)t * BATCH * HID);
    }

    // Final hidden state -> tail of d_out
    (void)hipMemcpyAsync(out + (size_t)MTOT * HID, h,
                         (size_t)BATCH * HID * sizeof(float),
                         hipMemcpyDeviceToDevice, stream);
}
